// DeformableHeadAttention_27023934226949
// MI455X (gfx1250) — compile-verified, matrix operand order checked by reading, not run on hardware
//
#include <hip/hip_runtime.h>

// ---------------------------------------------------------------------------
// CDNA5 / gfx1250 deformable attention.
// All five projections use v_wmma_f32_16x16x32_bf16 (bf16 in, f32 accum).
// Inner dim is always 256 (hardcoded -> shift/mask addressing, no int division).
// Bilinear sampling: one wave per (pixel, head); lane == dk channel (dk=32).
// ---------------------------------------------------------------------------

typedef __attribute__((ext_vector_type(16))) __bf16 v16bf;
typedef __attribute__((ext_vector_type(8)))  __bf16 v8bf;
typedef __attribute__((ext_vector_type(8)))  float  v8f;

#define KD 256   // shared inner dimension of every GEMM in this model

__device__ __forceinline__ unsigned short f2bf(float f) {
  union { float f; unsigned u; } v; v.f = f;
  unsigned u = v.u;
  u += 0x7FFFu + ((u >> 16) & 1u);           // round-to-nearest-even
  return (unsigned short)(u >> 16);
}

// ----------------------------- elementwise f32 -> bf16 ---------------------
__global__ void cvt_f32_bf16(const float* __restrict__ in,
                             unsigned short* __restrict__ out, long n) {
  long i = (long)blockIdx.x * blockDim.x + threadIdx.x;
  if (i < n) out[i] = f2bf(in[i]);
}

// ------------------- transpose (256 x N) f32 -> (N x 256) bf16 -------------
__global__ void tpose_f32_bf16(const float* __restrict__ W,
                               unsigned short* __restrict__ Wt, int N) {
  const int i = blockIdx.x * blockDim.x + threadIdx.x;
  if (i >= N * KD) return;
  const int n = i >> 8, k = i & 255;         // Kd == 256
  Wt[i] = f2bf(W[(long)k * N + n]);
}

// ---------------------------- WMMA bf16 GEMM -------------------------------
// C[M x N] = A[M x 256](bf16) * Bt[N x 256](bf16)^T + bias,
// M multiple of 128, N multiple of 64. Block = 4 waves; wave tile = 32 x 64
// (2 M-frags x 4 N-frags, A reuse x4). Fragment loads: two contiguous 16B
// runs per lane matching the 16-bit A/B WMMA VGPR layout (K groups
// {8h..8h+7} and {16+8h..16+8h+7}).
// MODE 0: outF[row*N+col] = v
// MODE 1: outBf[row*N+col] = bf16(v)
// MODE 2: head-permuted scatter kf[((row>>logS)*8 + col>>5) << logS | pix][col&31]
template <int MODE>
__global__ __launch_bounds__(128) void gemm_bf16_kernel(
    const unsigned short* __restrict__ A,
    const unsigned short* __restrict__ Bt,
    const float* __restrict__ bias,
    float* __restrict__ outF,
    unsigned short* __restrict__ outBf,
    int N, int logS) {
  const int lane = threadIdx.x & 31;
  const int wave = threadIdx.x >> 5;
  const int half = lane >> 4;
  const int l16  = lane & 15;
  const long m0  = (long)blockIdx.x * 128 + wave * 32;
  const int  n0  = blockIdx.y * 64;

  v8f c[2][4];
#pragma unroll
  for (int i = 0; i < 2; ++i)
#pragma unroll
    for (int j = 0; j < 4; ++j)
#pragma unroll
      for (int r = 0; r < 8; ++r) c[i][j][r] = 0.0f;

  for (int k0 = 0; k0 < KD; k0 += 32) {
    v16bf a[2], b[4];
#pragma unroll
    for (int i = 0; i < 2; ++i) {
      const unsigned short* p = A + ((m0 + i * 16 + l16) << 8) + k0 + half * 8;
      v8bf lo = *reinterpret_cast<const v8bf*>(p);
      v8bf hi = *reinterpret_cast<const v8bf*>(p + 16);
      a[i] = __builtin_shufflevector(lo, hi, 0,1,2,3,4,5,6,7,8,9,10,11,12,13,14,15);
    }
#pragma unroll
    for (int j = 0; j < 4; ++j) {
      const unsigned short* p = Bt + ((long)(n0 + j * 16 + l16) << 8) + k0 + half * 8;
      v8bf lo = *reinterpret_cast<const v8bf*>(p);
      v8bf hi = *reinterpret_cast<const v8bf*>(p + 16);
      b[j] = __builtin_shufflevector(lo, hi, 0,1,2,3,4,5,6,7,8,9,10,11,12,13,14,15);
    }
#pragma unroll
    for (int i = 0; i < 2; ++i)
#pragma unroll
      for (int j = 0; j < 4; ++j)
        c[i][j] = __builtin_amdgcn_wmma_f32_16x16x32_bf16(
            false, a[i], false, b[j], (short)0, c[i][j], false, false);
  }

#pragma unroll
  for (int i = 0; i < 2; ++i) {
#pragma unroll
    for (int j = 0; j < 4; ++j) {
      const int col = n0 + j * 16 + l16;
      const float bv = bias[col];
#pragma unroll
      for (int r = 0; r < 8; ++r) {
        const long row = m0 + i * 16 + half * 8 + r;
        const float v = c[i][j][r] + bv;
        if (MODE == 0) {
          outF[row * (long)N + col] = v;
        } else if (MODE == 1) {
          outBf[row * (long)N + col] = f2bf(v);
        } else {
          const long bb  = row >> logS;               // batch
          const long pix = row & ((1L << logS) - 1);  // y*wl + x
          const int head = col >> 5, d = col & 31;
          outF[((((bb * 8 + head) << logS) | pix) << 5) + d] = v;
        }
      }
    }
  }
}

// ------------------------- softmax over SCALES*K=16 ------------------------
__global__ void softmax16(float* __restrict__ a) {   // in-place, NQ x (8 x 16)
  long t = (long)blockIdx.x * blockDim.x + threadIdx.x;
  float* p = a + (t >> 3) * 128 + (t & 7) * 16;
  float m = p[0];
#pragma unroll
  for (int j = 1; j < 16; ++j) m = fmaxf(m, p[j]);
  float e[16], s = 0.f;
#pragma unroll
  for (int j = 0; j < 16; ++j) { e[j] = expf(p[j] - m); s += e[j]; }
  const float inv = 1.f / s;
#pragma unroll
  for (int j = 0; j < 16; ++j) p[j] = e[j] * inv;
}

// --------------------- bilinear sampling + weighted sum --------------------
// One wave per (b, pix, head); lane = dk channel. All address math is wave-
// uniform; each corner fetch is one coalesced 128B read of kf.
__global__ __launch_bounds__(256) void sample_kernel(
    const float* __restrict__ off,    // NQ x 256
    const float* __restrict__ attn,   // NQ x 128 (post-softmax)
    const float* __restrict__ kf,     // permuted key features, all scales
    const float* __restrict__ refpt,  // 64*64*2
    unsigned short* __restrict__ featBf) {
  const int gid  = blockIdx.x * 8 + (threadIdx.x >> 5);
  const int lane = threadIdx.x & 31;
  const int head = gid & 7;
  const int row  = gid >> 3;          // b*4096 + pix
  const int pix  = row & 4095;
  const int b    = row >> 12;

  const float rx = refpt[pix * 2 + 0];
  const float ry = refpt[pix * 2 + 1];
  const float* op = off  + (long)row * 256 + head * 32;
  const float* ap = attn + (long)row * 128 + head * 16;

  const int  wls[4]  = {16, 32, 64, 128};
  const long cumr[4] = {0, 1024, 5120, 21504};   // cumulative key rows

  float acc = 0.f;
#pragma unroll
  for (int l = 0; l < 4; ++l) {
    const int   wl   = wls[l], hl = wls[l], S = wl * hl;
    const float wlm1 = (float)(wl - 1), hlm1 = (float)(hl - 1);
    const float refx = rx * wlm1, refy = ry * hlm1;
    const float* ps  = kf + cumr[l] * 256 + ((long)(b * 8 + head) * S) * 32 + lane;
#pragma unroll
    for (int k = 0; k < 4; ++k) {
      const float ptx = refx + op[(l * 4 + k) * 2 + 0];
      const float pty = refy + op[(l * 4 + k) * 2 + 1];
      const float gx = 2.f * ptx / wlm1 - 1.f;
      const float gy = 2.f * pty / hlm1 - 1.f;
      const float ix = ((gx + 1.f) * wl - 1.f) * 0.5f;
      const float iy = ((gy + 1.f) * hl - 1.f) * 0.5f;
      const float x0 = floorf(ix), y0 = floorf(iy);
      const float wx1 = ix - x0, wy1 = iy - y0;
      const float wx0 = 1.f - wx1, wy0 = 1.f - wy1;

      auto samp = [&](float xx, float yy) -> float {
        if (xx < 0.f || xx > wlm1 || yy < 0.f || yy > hlm1) return 0.f;
        const int xc = (int)xx, yc = (int)yy;
        return ps[(long)(yc * wl + xc) * 32];
      };
      const float v = samp(x0,       y0)       * (wx0 * wy0)
                    + samp(x0 + 1.f, y0)       * (wx1 * wy0)
                    + samp(x0,       y0 + 1.f) * (wx0 * wy1)
                    + samp(x0 + 1.f, y0 + 1.f) * (wx1 * wy1);
      acc += ap[l * 4 + k] * v;
    }
  }
  // torch "raw view" order: flat (b, head, pix, dk) — viewed row-major as
  // (16384 x 256) this IS the reshaped operand of the final GEMM.
  featBf[((long)(b * 8 + head) * 4096 + pix) * 32 + lane] = f2bf(acc);
}

// ---------------------------------------------------------------------------
extern "C" void kernel_launch(void* const* d_in, const int* in_sizes, int n_in,
                              void* d_out, int out_size, void* d_ws, size_t ws_size,
                              hipStream_t stream) {
  (void)in_sizes; (void)n_in; (void)out_size; (void)ws_size;
  const float* query = (const float*)d_in[0];
  const float* refpt = (const float*)d_in[1];
  const float* Wq    = (const float*)d_in[2];
  const float* bq    = (const float*)d_in[3];
  const float* Wk    = (const float*)d_in[4];
  const float* bk    = (const float*)d_in[5];
  const float* Woff  = (const float*)d_in[6];
  const float* boff  = (const float*)d_in[7];
  const float* WA    = (const float*)d_in[8];
  const float* bA    = (const float*)d_in[9];
  const float* Wm    = (const float*)d_in[10];
  const float* bm    = (const float*)d_in[11];
  const float* keys[4] = {(const float*)d_in[12], (const float*)d_in[13],
                          (const float*)d_in[14], (const float*)d_in[15]};

  const long NQ = 16384;                        // 4 * 64 * 64
  const int  rowsL[4] = {1024, 4096, 16384, 65536};
  const int  logSL[4] = {8, 10, 12, 14};        // log2(wl*hl)
  const long cumr[4]  = {0, 1024, 5120, 21504};

  // workspace layout (all offsets 256B-aligned)
  char* ws = (char*)d_ws;
  const size_t o_xbf   = 0;                       // 16384*256 bf16
  const size_t o_wqt   = o_xbf   + 8388608;
  const size_t o_wkt   = o_wqt   + 131072;
  const size_t o_wofft = o_wkt   + 131072;
  const size_t o_wmt   = o_wofft + 131072;
  const size_t o_wat   = o_wmt   + 131072;        // 128*256 bf16
  const size_t o_qbf   = o_wat   + 65536;         // 16384*256 bf16
  const size_t o_off   = o_qbf   + 8388608;       // 16384*256 f32
  const size_t o_log   = o_off   + 16777216;      // 16384*128 f32
  const size_t o_kbf   = o_log   + 8388608;       // 65536*256 bf16 (reused)
  const size_t o_kf    = o_kbf   + 33554432;      // 87040*256 f32
  const size_t o_fbf   = o_kf    + 89128960;      // 16384*256 bf16

  unsigned short* xbf   = (unsigned short*)(ws + o_xbf);
  unsigned short* wqt   = (unsigned short*)(ws + o_wqt);
  unsigned short* wkt   = (unsigned short*)(ws + o_wkt);
  unsigned short* wofft = (unsigned short*)(ws + o_wofft);
  unsigned short* wmt   = (unsigned short*)(ws + o_wmt);
  unsigned short* wat   = (unsigned short*)(ws + o_wat);
  unsigned short* qbf   = (unsigned short*)(ws + o_qbf);
  float*          offp  = (float*)(ws + o_off);
  float*          logit = (float*)(ws + o_log);
  unsigned short* kbf   = (unsigned short*)(ws + o_kbf);
  float*          kf    = (float*)(ws + o_kf);
  unsigned short* fbf   = (unsigned short*)(ws + o_fbf);

  // 1) activations + weights -> bf16 (weights transposed to N x K)
  cvt_f32_bf16<<<16384, 256, 0, stream>>>(query, xbf, NQ * 256);
  tpose_f32_bf16<<<256, 256, 0, stream>>>(Wq,   wqt,   256);
  tpose_f32_bf16<<<256, 256, 0, stream>>>(Wk,   wkt,   256);
  tpose_f32_bf16<<<256, 256, 0, stream>>>(Woff, wofft, 256);
  tpose_f32_bf16<<<256, 256, 0, stream>>>(Wm,   wmt,   256);
  tpose_f32_bf16<<<128, 256, 0, stream>>>(WA,   wat,   128);

  // 2) q = x @ Wq + bq  (bf16 result only; consumed by two more GEMMs)
  gemm_bf16_kernel<1><<<dim3(128, 4), 128, 0, stream>>>(
      xbf, wqt, bq, nullptr, qbf, 256, 0);
  // 3) offsets = q @ Woff + boff ; logits = q @ WA + bA
  gemm_bf16_kernel<0><<<dim3(128, 4), 128, 0, stream>>>(
      qbf, wofft, boff, offp, nullptr, 256, 0);
  gemm_bf16_kernel<0><<<dim3(128, 2), 128, 0, stream>>>(
      qbf, wat, bA, logit, nullptr, 128, 0);
  softmax16<<<512, 256, 0, stream>>>(logit);

  // 4) key projections per scale, scattered straight into (b,head,y,x,dk)
  for (int l = 0; l < 4; ++l) {
    const long cnt = (long)rowsL[l] * 256;
    cvt_f32_bf16<<<(unsigned)((cnt + 255) / 256), 256, 0, stream>>>(keys[l], kbf, cnt);
    gemm_bf16_kernel<2><<<dim3(rowsL[l] / 128, 4), 128, 0, stream>>>(
        kbf, wkt, bk, kf + cumr[l] * 256, nullptr, 256, logSL[l]);
  }

  // 5) bilinear sampling + attention-weighted sum -> feat (bf16, raw-view order)
  sample_kernel<<<16384, 256, 0, stream>>>(offp, logit, kf, refpt, fbf);

  // 6) out = feat @ Wm + bm  (f32 to d_out)
  gemm_bf16_kernel<0><<<dim3(128, 4), 128, 0, stream>>>(
      fbf, wmt, bm, (float*)d_out, nullptr, 256, 0);
}
